// GroupedQueryAttention_17265768530559
// MI455X (gfx1250) — compile-verified, matrix operand order checked by reading, not run on hardware
//
#include <hip/hip_runtime.h>
#include <math.h>

// ---------------- problem constants ----------------
#define B_   2
#define S_   2048
#define DIN  2048
#define H_   16
#define G_   4
#define HD_  128
#define GS_  (H_ / G_)      // 4 heads per kv group
#define NQ   (H_ * HD_)     // 2048
#define NKV  (G_ * HD_)     // 512
#define BS   (B_ * S_)      // 4096
#define SCALE_F 0.08838834764831845f  // 128^-0.5

typedef __attribute__((ext_vector_type(16))) __bf16 v16bf;
typedef __attribute__((ext_vector_type(8)))  __bf16 v8bf;
typedef __attribute__((ext_vector_type(8)))  float  v8f;

// ---------------- CDNA5 async global->LDS copy (16B per lane) -------------
__device__ inline void async_copy_b128(void* lds_dst, const void* gsrc) {
  unsigned lds = (unsigned)(size_t)lds_dst;   // LDS aperture: addr[31:0] = LDS offset
  unsigned long long ga = (unsigned long long)(size_t)gsrc;
  asm volatile("global_load_async_to_lds_b128 %0, %1, off"
               :: "v"(lds), "v"(ga) : "memory");
}
__device__ inline void async_wait0() {
#if __has_builtin(__builtin_amdgcn_s_wait_asynccnt)
  __builtin_amdgcn_s_wait_asynccnt(0);
#else
  asm volatile("s_wait_asynccnt 0x0" ::: "memory");
#endif
}

// 16-element bf16 fragment per lane, CDNA5 16-bit A/B layout
// (two runs of 8 K-values per lane).
__device__ inline v16bf load_frag16(const __bf16* base, int lh) {
  const v8bf lo = *(const v8bf*)(base + lh * 8);
  const v8bf hi = *(const v8bf*)(base + 16 + lh * 8);
  v16bf r;
#pragma unroll
  for (int i = 0; i < 8; ++i) { r[i] = lo[i]; r[i + 8] = hi[i]; }
  return r;
}

__device__ inline v8f wmma_bf16(v16bf a, v16bf b, v8f c) {
  return __builtin_amdgcn_wmma_f32_16x16x32_bf16(false, a, false, b, (short)0,
                                                 c, false, false);
}

// ---------------- f32 -> bf16 convert (x only) ----------------
__global__ void cvt_bf16_kernel(const float* __restrict__ in,
                                __bf16* __restrict__ out, int n) {
  int i4 = (blockIdx.x * blockDim.x + threadIdx.x) * 4;
  if (i4 + 3 < n) {
    float4 f = *(const float4*)(in + i4);
    out[i4 + 0] = (__bf16)f.x;
    out[i4 + 1] = (__bf16)f.y;
    out[i4 + 2] = (__bf16)f.z;
    out[i4 + 3] = (__bf16)f.w;
  } else {
    for (int j = i4; j < n; ++j) out[j] = (__bf16)in[j];
  }
}

// ---------------- f32 [K][N] -> bf16 transposed [N][K] (weights, one-time)
__launch_bounds__(256)
__global__ void cvtT_kernel(const float* __restrict__ in,
                            __bf16* __restrict__ out, int K, int N) {
  __shared__ float t[32][33];
  const int k0 = blockIdx.y * 32, n0 = blockIdx.x * 32;
  const int tx = threadIdx.x & 31, ty = threadIdx.x >> 5;   // 32 x 8
#pragma unroll
  for (int i = 0; i < 32; i += 8)
    t[ty + i][tx] = in[(size_t)(k0 + ty + i) * N + n0 + tx];
  __syncthreads();
#pragma unroll
  for (int i = 0; i < 32; i += 8)
    out[(size_t)(n0 + ty + i) * K + k0 + tx] = (__bf16)t[tx][ty + i];
}

// ---------------- vproj f32 [b*s][g*HD+d] -> vbT bf16 [(b*G+g)*HD+d][s] ---
__launch_bounds__(256)
__global__ void vT_kernel(const float* __restrict__ in, __bf16* __restrict__ out) {
  __shared__ float t[32][33];
  const int b = blockIdx.z;
  const int s0 = blockIdx.y * 32, c0 = blockIdx.x * 32;
  const int tx = threadIdx.x & 31, ty = threadIdx.x >> 5;
#pragma unroll
  for (int i = 0; i < 32; i += 8)
    t[ty + i][tx] = in[((size_t)(b * S_) + s0 + ty + i) * NKV + c0 + tx];
  __syncthreads();
#pragma unroll
  for (int i = 0; i < 32; i += 8) {
    int col = c0 + ty + i, g = col >> 7, d = col & 127;
    out[(((size_t)(b * G_ + g)) * HD_ + d) * S_ + s0 + tx] = (__bf16)t[tx][ty + i];
  }
}

// ---------------- bf16 WMMA GEMM: C(f32) = A(bf16) @ BT(bf16)^T ----------
// A: [M][K] row-major, BT: [N][K] row-major (pre-transposed weights).
// Block: 128 threads = 4 waves. Block tile 128(M) x 128(N); each wave owns
// 32 rows (2 m-tiles) x 128 cols -> each B fragment feeds 2 WMMAs.
// K-step 64, double-buffered LDS staged via async global->LDS copies,
// fragment loads software-pipelined one tile ahead.
__launch_bounds__(128)
__global__ void gemm_bf16_nt(const __bf16* __restrict__ A,
                             const __bf16* __restrict__ BT,
                             float* __restrict__ C, int M, int N, int K) {
  __shared__ __bf16 As[2][128 * 64];   // [m][k]
  __shared__ __bf16 Bs[2][128 * 64];   // [n][k]
  const int tid = threadIdx.x, wave = tid >> 5, lane = tid & 31;
  const int lh = lane >> 4, ll = lane & 15;
  const int m0 = blockIdx.y * 128, n0 = blockIdx.x * 128;
  const int nk = K >> 6;

  v8f acc0[8] = {};
  v8f acc1[8] = {};

  auto stage = [&](int buf, int k0) {
#pragma unroll
    for (int i = 0; i < 8; ++i) {           // A tile: 128x64
      int idx = (tid + i * 128) * 8;
      int r = idx >> 6, c = idx & 63;
      async_copy_b128(&As[buf][r * 64 + c],
                      &A[(size_t)(m0 + r) * K + k0 + c]);
    }
#pragma unroll
    for (int i = 0; i < 8; ++i) {           // B tile: 128x64
      int idx = (tid + i * 128) * 8;
      int n = idx >> 6, c = idx & 63;
      async_copy_b128(&Bs[buf][n * 64 + c],
                      &BT[(size_t)(n0 + n) * K + k0 + c]);
    }
  };

  stage(0, 0);
  for (int kt = 0; kt < nk; ++kt) {
    const int cur = kt & 1;
    async_wait0();
    __syncthreads();
    if (kt + 1 < nk) stage(cur ^ 1, (kt + 1) * 64);
#pragma unroll
    for (int kk = 0; kk < 64; kk += 32) {
      v16bf a0 = load_frag16(&As[cur][(wave * 32 + ll) * 64 + kk], lh);
      v16bf a1 = load_frag16(&As[cur][(wave * 32 + 16 + ll) * 64 + kk], lh);
      v16bf b  = load_frag16(&Bs[cur][ll * 64 + kk], lh);
#pragma unroll
      for (int t = 0; t < 8; ++t) {
        v16bf bn = b;
        if (t < 7) bn = load_frag16(&Bs[cur][((t + 1) * 16 + ll) * 64 + kk], lh);
        acc0[t] = wmma_bf16(a0, b, acc0[t]);
        acc1[t] = wmma_bf16(a1, b, acc1[t]);
        b = bn;
      }
    }
  }
  const int ncol = n0 + ll;
#pragma unroll
  for (int t = 0; t < 8; ++t) {
#pragma unroll
    for (int r = 0; r < 8; ++r) {
      int m_a = m0 + wave * 32 + lh * 8 + r;
      int m_b = m_a + 16;
      C[(size_t)m_a * N + ncol + t * 16] = acc0[t][r];
      C[(size_t)m_b * N + ncol + t * 16] = acc1[t][r];
    }
  }
}

// ---------------- RMSNorm + RoPE (+scale), f32 proj -> bf16 ----------------
__launch_bounds__(128)
__global__ void qk_postproc(const float* __restrict__ proj,
                            __bf16* __restrict__ out,
                            const float* __restrict__ normw,
                            const float* __restrict__ cosT,
                            const float* __restrict__ sinT,
                            int nheads, float scale) {
  const int d = threadIdx.x;                  // 0..127
  const int h = blockIdx.x % nheads;
  const int bs = blockIdx.x / nheads;
  const int s = bs % S_;
  const size_t base = (size_t)bs * ((size_t)nheads * HD_) + (size_t)h * HD_;

  __shared__ float xsh[HD_];
  __shared__ float red[128];

  float x = proj[base + d];
  red[d] = x * x;
  __syncthreads();
#pragma unroll
  for (int offp = 64; offp > 0; offp >>= 1) {
    if (d < offp) red[d] += red[d + offp];
    __syncthreads();
  }
  float rr = rsqrtf(red[0] * (1.0f / HD_) + 1e-6f);
  float xn = x * rr * normw[d];
  xsh[d] = xn;
  __syncthreads();
  float rot = (d < HD_ / 2) ? -xsh[d + HD_ / 2] : xsh[d - HD_ / 2];
  float o = (xn * cosT[(size_t)s * HD_ + d] + rot * sinT[(size_t)s * HD_ + d]) * scale;
  out[base + d] = (__bf16)o;
}

// ---------------- flash attention (causal, GQA) ----------------
// Grid: (B*H, S/64). Block: 128 threads = 4 waves; wave w owns query rows
// [16w,16w+16). K/V tiles double-buffered via async copies; fragment loads
// pipelined one tile ahead.
__launch_bounds__(128)
__global__ void flash_attn(const __bf16* __restrict__ qb,
                           const __bf16* __restrict__ kb,
                           const __bf16* __restrict__ vbT,
                           __bf16* __restrict__ ctx) {
  const int tid = threadIdx.x, wave = tid >> 5, lane = tid & 31;
  const int lh = lane >> 4, ll = lane & 15;
  const int bh = blockIdx.x;
  const int b = bh / H_, h = bh % H_, g = h / GS_;
  const int i0 = blockIdx.y * 64;

  __shared__ __bf16 Qs[64 * HD_];      // [q][d]
  __shared__ __bf16 Ks[2][64 * HD_];   // [k][d]
  __shared__ __bf16 Vt[2][HD_ * 64];   // [d][k]
  __shared__ float  Ss[64 * 64];
  __shared__ __bf16 Ps[64 * 64];
  __shared__ float  Lm[64], Ll[64], La[64];

  const size_t vbase = ((size_t)(b * G_ + g)) * HD_ * S_;

  auto stageKV = [&](int buf, int j0) {
#pragma unroll
    for (int i = 0; i < 8; ++i) {        // K tile [key][d]
      int idx = (tid + i * 128) * 8;
      int r = idx >> 7, c = idx & 127;
      async_copy_b128(&Ks[buf][r * HD_ + c],
                      &kb[((size_t)(b * S_ + j0 + r)) * NKV + g * HD_ + c]);
    }
#pragma unroll
    for (int i = 0; i < 8; ++i) {        // V^T tile [d][key] (pre-transposed)
      int idx = (tid + i * 128) * 8;
      int d = idx >> 6, c = idx & 63;
      async_copy_b128(&Vt[buf][d * 64 + c],
                      &vbT[vbase + (size_t)d * S_ + j0 + c]);
    }
  };

  // prologue: Q tile + first K/V tile, all async
#pragma unroll
  for (int i = 0; i < 8; ++i) {
    int idx = (tid + i * 128) * 8;
    int r = idx >> 7, c = idx & 127;
    async_copy_b128(&Qs[r * HD_ + c],
                    &qb[((size_t)(b * S_ + i0 + r)) * NQ + h * HD_ + c]);
  }
  stageKV(0, 0);
  if (tid < 64) { Lm[tid] = -3.0e38f; Ll[tid] = 0.0f; }

  v8f oacc[8] = {};
  const int nj = (i0 >> 6) + 1;

  for (int jt = 0; jt < nj; ++jt) {
    const int j0 = jt * 64, cur = jt & 1;
    async_wait0();
    __syncthreads();
    if (jt + 1 < nj) stageKV(cur ^ 1, j0 + 64);

    // S = Q @ K^T : each wave 16x64 (4 tiles), contraction over HD_=128
    v8f sacc[4] = {};
#pragma unroll
    for (int kk = 0; kk < HD_; kk += 32) {
      v16bf a = load_frag16(&Qs[(wave * 16 + ll) * HD_ + kk], lh);
      v16bf bfr = load_frag16(&Ks[cur][ll * HD_ + kk], lh);
#pragma unroll
      for (int t = 0; t < 4; ++t) {
        v16bf bn = bfr;
        if (t < 3) bn = load_frag16(&Ks[cur][((t + 1) * 16 + ll) * HD_ + kk], lh);
        sacc[t] = wmma_bf16(a, bfr, sacc[t]);
        bfr = bn;
      }
    }
    {
      int mrow = wave * 16 + lh * 8;
#pragma unroll
      for (int t = 0; t < 4; ++t)
#pragma unroll
        for (int r = 0; r < 8; ++r)
          Ss[(mrow + r) * 64 + t * 16 + ll] = sacc[t][r];
    }
    __syncthreads();

    // online softmax: one thread per row (causal mask analytic)
    if (tid < 64) {
      const int row = tid, qpos = i0 + row;
      float mprev = Lm[row];
      float mx = mprev;
      for (int c = 0; c < 64; ++c) {
        float sc = Ss[row * 64 + c];
        if (j0 + c > qpos) sc = -3.0e38f;
        mx = fmaxf(mx, sc);
      }
      float alpha = __expf(mprev - mx);
      float lsum = 0.0f;
      for (int c = 0; c < 64; ++c) {
        float sc = Ss[row * 64 + c];
        if (j0 + c > qpos) sc = -3.0e38f;
        float p = __expf(sc - mx);
        lsum += p;
        Ps[row * 64 + c] = (__bf16)p;
      }
      Lm[row] = mx;
      Ll[row] = Ll[row] * alpha + lsum;
      La[row] = alpha;
    }
    __syncthreads();

    // rescale O by alpha(row), then O += P @ V
    {
      int mbase = wave * 16 + lh * 8;
#pragma unroll
      for (int r = 0; r < 8; ++r) {
        float al = La[mbase + r];
#pragma unroll
        for (int t = 0; t < 8; ++t) oacc[t][r] *= al;
      }
    }
#pragma unroll
    for (int kc = 0; kc < 64; kc += 32) {
      v16bf a = load_frag16(&Ps[(wave * 16 + ll) * 64 + kc], lh);
      v16bf bfr = load_frag16(&Vt[cur][ll * 64 + kc], lh);
#pragma unroll
      for (int t = 0; t < 8; ++t) {
        v16bf bn = bfr;
        if (t < 7) bn = load_frag16(&Vt[cur][((t + 1) * 16 + ll) * 64 + kc], lh);
        oacc[t] = wmma_bf16(a, bfr, oacc[t]);
        bfr = bn;
      }
    }
  }
  __syncthreads();

  // epilogue: divide by l, store ctx bf16 at [b*S+s][h*HD + d]
  {
    int mbase = wave * 16 + lh * 8;
#pragma unroll
    for (int r = 0; r < 8; ++r) {
      float linv = 1.0f / Ll[mbase + r];
      int srow = i0 + mbase + r;
#pragma unroll
      for (int t = 0; t < 8; ++t)
        ctx[((size_t)(b * S_ + srow)) * NQ + h * HD_ + t * 16 + ll] =
            (__bf16)(oacc[t][r] * linv);
    }
  }
}

// ---------------- host launcher ----------------
extern "C" void kernel_launch(void* const* d_in, const int* in_sizes, int n_in,
                              void* d_out, int out_size, void* d_ws, size_t ws_size,
                              hipStream_t stream) {
  const float* x    = (const float*)d_in[0];
  // d_in[1] = mask (unused; causal mask computed analytically)
  const float* cosT = (const float*)d_in[2];
  const float* sinT = (const float*)d_in[3];
  const float* Wq   = (const float*)d_in[4];
  const float* Wk   = (const float*)d_in[5];
  const float* Wv   = (const float*)d_in[6];
  const float* Wo   = (const float*)d_in[7];
  const float* qnw  = (const float*)d_in[8];
  const float* knw  = (const float*)d_in[9];

  char* ws = (char*)d_ws;
  size_t off = 0;
  auto alloc = [&](size_t bytes) -> char* {
    char* p = ws + off;
    off = (off + bytes + 255) & ~(size_t)255;
    return p;
  };

  __bf16* xb    = (__bf16*)alloc((size_t)BS * DIN * 2);
  __bf16* wqT   = (__bf16*)alloc((size_t)DIN * NQ * 2);   // [NQ][DIN]
  __bf16* wkT   = (__bf16*)alloc((size_t)DIN * NKV * 2);  // [NKV][DIN]
  __bf16* wvT   = (__bf16*)alloc((size_t)DIN * NKV * 2);  // [NKV][DIN]
  __bf16* woT   = (__bf16*)alloc((size_t)NQ * DIN * 2);   // [DIN][NQ]
  float*  qproj = (float*)alloc((size_t)BS * NQ * 4);
  float*  kproj = (float*)alloc((size_t)BS * NKV * 4);
  float*  vproj = (float*)alloc((size_t)BS * NKV * 4);
  __bf16* qb    = (__bf16*)alloc((size_t)BS * NQ * 2);
  __bf16* kb    = (__bf16*)alloc((size_t)BS * NKV * 2);
  __bf16* vbT   = (__bf16*)alloc((size_t)BS * NKV * 2);   // [(b*G+g)*HD+d][s]
  __bf16* ctxb  = (__bf16*)qproj;  // reuse: qproj consumed before attention

  // 1) bf16 conversions: x straight, weights transposed
  {
    int n = BS * DIN;
    cvt_bf16_kernel<<<(n / 4 + 255) / 256, 256, 0, stream>>>(x, xb, n);
  }
  cvtT_kernel<<<dim3(NQ / 32,  DIN / 32), 256, 0, stream>>>(Wq, wqT, DIN, NQ);
  cvtT_kernel<<<dim3(NKV / 32, DIN / 32), 256, 0, stream>>>(Wk, wkT, DIN, NKV);
  cvtT_kernel<<<dim3(NKV / 32, DIN / 32), 256, 0, stream>>>(Wv, wvT, DIN, NKV);
  cvtT_kernel<<<dim3(DIN / 32, NQ / 32),  256, 0, stream>>>(Wo, woT, NQ, DIN);

  // 2) QKV projections (WMMA bf16 -> f32), async-staged, double-buffered
  gemm_bf16_nt<<<dim3(NQ / 128,  BS / 128), 128, 0, stream>>>(xb, wqT, qproj, BS, NQ,  DIN);
  gemm_bf16_nt<<<dim3(NKV / 128, BS / 128), 128, 0, stream>>>(xb, wkT, kproj, BS, NKV, DIN);
  gemm_bf16_nt<<<dim3(NKV / 128, BS / 128), 128, 0, stream>>>(xb, wvT, vproj, BS, NKV, DIN);

  // 3) RMSNorm + RoPE (+scale for q); v: transpose-convert per (b,g)
  qk_postproc<<<BS * H_, 128, 0, stream>>>(qproj, qb, qnw, cosT, sinT, H_, SCALE_F);
  qk_postproc<<<BS * G_, 128, 0, stream>>>(kproj, kb, knw, cosT, sinT, G_, 1.0f);
  vT_kernel<<<dim3(NKV / 32, S_ / 32, B_), 256, 0, stream>>>(vproj, vbT);

  // 4) causal GQA flash attention (WMMA, async-staged K/V, double-buffered)
  flash_attn<<<dim3(B_ * H_, S_ / 64), 128, 0, stream>>>(qb, kb, vbT, ctxb);

  // 5) output projection -> f32 d_out
  gemm_bf16_nt<<<dim3(DIN / 128, BS / 128), 128, 0, stream>>>(ctxb, woT, (float*)d_out, BS, DIN, NQ);

  (void)in_sizes; (void)n_in; (void)out_size; (void)ws_size;
}
